// LSTMAgent_57054345560439
// MI455X (gfx1250) — matrix operand order repaired and checked by reading, not compiled, not run on hardware
//
#include <hip/hip_runtime.h>
#include <hip/hip_bf16.h>

typedef __attribute__((ext_vector_type(16))) _Float16 v16h;
typedef __attribute__((ext_vector_type(8)))  float    v8f;
typedef int v4i __attribute__((vector_size(16)));

typedef __attribute__((address_space(1))) v4i* gp_v4i;   // global int4*
typedef __attribute__((address_space(3))) v4i* lp_v4i;   // LDS int4*

#if __has_builtin(__builtin_amdgcn_global_load_async_to_lds_b128)
#define HAS_ASYNC_LDS 1
#else
#define HAS_ASYNC_LDS 0
#endif

#define TB    8192
#define NCH   26
#define HW    256
#define FEAT  6656
#define HID   512
#define G4    2048
#define BSZ   128
#define TSTEP 64
#define NACT  6

// ---------------------------------------------------------------------------
// Weight packing: torch weight src[Nsrc][Ksrc] (out x in) -> B-fragment tiles.
// Fragment (kt,nt): 32 lanes x 16 halves. Per ISA 16-bit B 32x16 layout:
//   col n = nt*16 + (lane&15), k = kt*32 + (lane>>4)*8 + (h < 8 ? h : h + 8)
__global__ void pack_b(const float* __restrict__ src, _Float16* __restrict__ dst,
                       int Nsrc, int Ksrc, int NT) {
    int frag = blockIdx.x;            // kt*NT + nt
    int lane = threadIdx.x;           // 0..31
    int kt = frag / NT, nt = frag % NT;
    int n = nt * 16 + (lane & 15);
    int kbase = kt * 32 + (lane >> 4) * 8;
    _Float16* out = dst + ((size_t)frag * 32 + lane) * 16;
#pragma unroll
    for (int h = 0; h < 16; ++h) {
        int k = kbase + (h < 8 ? h : h + 8);
        float v = (n < Nsrc && k < Ksrc) ? src[(size_t)n * Ksrc + k] : 0.f;
        out[h] = (_Float16)v;
    }
}

__global__ void add_vec(float* __restrict__ dst, const float* __restrict__ a,
                        const float* __restrict__ b, int n) {
    int i = blockIdx.x * blockDim.x + threadIdx.x;
    if (i < n) dst[i] = a[i] + b[i];
}

__global__ void cvt_f16(_Float16* __restrict__ dst, const float* __restrict__ src, int n) {
    int i = blockIdx.x * blockDim.x + threadIdx.x;
    if (i < n) dst[i] = (_Float16)src[i];
}

__global__ void copy_f32(float* __restrict__ dst, const float* __restrict__ src, int n) {
    int i = blockIdx.x * blockDim.x + threadIdx.x;
    if (i < n) dst[i] = src[i];
}

// ---------------------------------------------------------------------------
// Implicit-GEMM conv via WMMA. One block = one image; 16 waves = 16 m-tiles
// (256 pixels); each wave computes BOTH 16-wide out-channel tiles so every
// im2col A gather feeds 2 WMMAs.
template <int KSZ>
__global__ __launch_bounds__(512) void conv_wmma(
    const _Float16* __restrict__ in,   // [N][26][16][16] f16
    const _Float16* __restrict__ Wp,   // packed B fragments, KT x 2
    const float* __restrict__ bias,    // [26]
    _Float16* __restrict__ outp)       // [N][26][16][16] f16
{
    constexpr int P  = KSZ / 2;
    constexpr int PD = 16 + 2 * P;
    __shared__ _Float16 img[NCH * PD * PD];

    const int n   = blockIdx.x;
    const int tid = threadIdx.x;

    for (int i = tid; i < NCH * PD * PD; i += 512) img[i] = (_Float16)0.f;
    __syncthreads();
    const _Float16* src = in + (size_t)n * NCH * HW;
    for (int i = tid; i < NCH * HW; i += 512) {
        int c = i >> 8, hw = i & 255, h = hw >> 4, w = hw & 15;
        img[(c * PD + (h + P)) * PD + (w + P)] = src[i];
    }
    __syncthreads();

    const int wave = tid >> 5, lane = tid & 31;
    const int side = lane >> 4;
    const int m  = wave * 16 + (lane & 15);
    const int oh = m >> 4, ow = m & 15;
    constexpr int Ksrc = NCH * KSZ * KSZ;
    constexpr int KT = (Ksrc + 31) >> 5;

    v8f acc0 = {}, acc1 = {};
    for (int kt = 0; kt < KT; ++kt) {
        v16h a;
#pragma unroll
        for (int h = 0; h < 16; ++h) {
            int k = kt * 32 + (h < 8 ? h : h + 8) + side * 8;
            _Float16 v = (_Float16)0.f;
            if (k < Ksrc) {
                int ci = k / (KSZ * KSZ), rem = k % (KSZ * KSZ);
                int kh = rem / KSZ, kw = rem % KSZ;
                v = img[(ci * PD + oh + kh) * PD + (ow + kw)];
            }
            a[h] = v;
        }
        const _Float16* wbase = Wp + (size_t)kt * 2 * 512 + lane * 16;
        v16h b0 = *(const v16h*)wbase;
        v16h b1 = *(const v16h*)(wbase + 512);
        acc0 = __builtin_amdgcn_wmma_f32_16x16x32_f16(false, a, false, b0,
                                                      (short)0, acc0, false, false);
        acc1 = __builtin_amdgcn_wmma_f32_16x16x32_f16(false, a, false, b1,
                                                      (short)0, acc1, false, false);
    }

#pragma unroll
    for (int ntl = 0; ntl < 2; ++ntl) {
        int co = ntl * 16 + (lane & 15);
        if (co < NCH) {
            float bb = bias[co];
#pragma unroll
            for (int r = 0; r < 8; ++r) {
                int pix = wave * 16 + side * 8 + r;
                float y = (ntl ? acc1[r] : acc0[r]) + bb;
                y = y > 0.f ? y : 0.01f * y;                  // LeakyReLU
                outp[((size_t)n * NCH + co) * HW + pix] = (_Float16)y;
            }
        }
    }
}

// ---------------------------------------------------------------------------
// Generic WMMA GEMM: Y = act( X[M,K] @ Wp(K,Npad packed) + bias + Cadd )
// 256 threads / 8 waves; block tile 128x32; each wave owns one m-tile and
// both n-tiles (A reuse -> 2 WMMA per A fragment). K-loop double-buffers the
// X tile in LDS; staging uses GLOBAL_LOAD_ASYNC_TO_LDS_B128 when available.
__global__ __launch_bounds__(256) void gemm_wmma(
    const _Float16* __restrict__ X, const _Float16* __restrict__ Wp,
    const float* __restrict__ bias, const float* __restrict__ Cadd, int ldc,
    float* __restrict__ Yf32, int ldy, int nvalid,
    _Float16* __restrict__ Yf16, int ldh,
    int M, int Npad, int K, int leaky)
{
    __shared__ __align__(16) _Float16 xs[2][128 * 32];
    const int tid  = threadIdx.x;
    const int wave = tid >> 5, lane = tid & 31;
    const int m0   = blockIdx.x * 128;
    const int nb0  = blockIdx.y * 32;
    const int NT   = Npad >> 4;
    const int side = lane >> 4;
    const int mrow = wave * 16 + (lane & 15);   // A row within the 128-row tile
    const int KT   = K >> 5;

    const int srow = tid >> 1;          // staging row 0..127
    const int sch  = (tid & 1) * 16;    // staging chunk of 16 halves (32B)

    auto stage = [&](int kt, int buf) {
        const _Float16* g = X + (size_t)(m0 + srow) * K + kt * 32 + sch;
        _Float16* l = &xs[buf][srow * 32 + sch];
#if HAS_ASYNC_LDS
        __builtin_amdgcn_global_load_async_to_lds_b128((gp_v4i)g, (lp_v4i)l, 0, 0);
        __builtin_amdgcn_global_load_async_to_lds_b128((gp_v4i)(g + 8),
                                                       (lp_v4i)(l + 8), 0, 0);
#else
        *(uint4*)l       = *(const uint4*)g;
        *(uint4*)(l + 8) = *(const uint4*)(g + 8);
        __builtin_prefetch(g + 32, 0, 1);
#endif
    };

    v8f acc0 = {}, acc1 = {};
    stage(0, 0);
    for (int kt = 0; kt < KT; ++kt) {
        const int cur = kt & 1;
#if HAS_ASYNC_LDS
#if __has_builtin(__builtin_amdgcn_s_wait_asynccnt)
        __builtin_amdgcn_s_wait_asynccnt(0);
#else
        asm volatile("s_wait_asynccnt 0x0" ::: "memory");
#endif
#endif
        __syncthreads();                 // all waves' staging of `cur` visible
        if (kt + 1 < KT) stage(kt + 1, cur ^ 1);

        // A fragment per ISA 16-bit A 16x32 lane/K map (2 contiguous 8-half runs)
        v16h a;
#pragma unroll
        for (int h = 0; h < 16; ++h) {
            int kk = (h < 8 ? h : h + 8) + side * 8;
            a[h] = xs[cur][mrow * 32 + kk];
        }
        // two B fragments: contiguous 32B per lane each
        const _Float16* wbase = Wp + ((size_t)kt * NT + (nb0 >> 4)) * 512 + lane * 16;
        v16h b0 = *(const v16h*)wbase;
        v16h b1 = *(const v16h*)(wbase + 512);
        acc0 = __builtin_amdgcn_wmma_f32_16x16x32_f16(false, a, false, b0,
                                                      (short)0, acc0, false, false);
        acc1 = __builtin_amdgcn_wmma_f32_16x16x32_f16(false, a, false, b1,
                                                      (short)0, acc1, false, false);
    }

#pragma unroll
    for (int ntl = 0; ntl < 2; ++ntl) {
        int n = nb0 + ntl * 16 + (lane & 15);
#pragma unroll
        for (int r = 0; r < 8; ++r) {
            int m = m0 + wave * 16 + side * 8 + r;
            float y = ntl ? acc1[r] : acc0[r];
            if (bias && n < nvalid) y += bias[n];
            if (Cadd) y += Cadd[(size_t)m * ldc + n];
            if (leaky) y = y > 0.f ? y : 0.01f * y;
            if (Yf32 && n < nvalid) Yf32[(size_t)m * ldy + n] = y;
            if (Yf16) Yf16[(size_t)m * ldh + n] = (_Float16)y;
        }
    }
}

// ---------------------------------------------------------------------------
// LSTM pointwise kernels (128 x 512 elements each)
__global__ void lstm_mask(float* __restrict__ h, float* __restrict__ c,
                          _Float16* __restrict__ h16, const float* __restrict__ done_t) {
    int i = blockIdx.x * blockDim.x + threadIdx.x;
    if (i >= BSZ * HID) return;
    int b = i >> 9;
    float m = 1.f - done_t[b];
    float hv = h[i] * m;
    h[i] = hv;
    h16[i] = (_Float16)hv;
    c[i] *= m;
}

__global__ void lstm_gates(const float* __restrict__ G, float* __restrict__ c,
                           float* __restrict__ h, _Float16* __restrict__ h16,
                           _Float16* __restrict__ hs16) {
    int i = blockIdx.x * blockDim.x + threadIdx.x;
    if (i >= BSZ * HID) return;
    int b = i >> 9, j = i & (HID - 1);
    const float* g = G + (size_t)b * G4;
    float ig = 1.f / (1.f + __expf(-g[j]));
    float fg = 1.f / (1.f + __expf(-g[HID + j]));
    float gg = tanhf(g[2 * HID + j]);
    float og = 1.f / (1.f + __expf(-g[3 * HID + j]));
    float cv = fg * c[i] + ig * gg;
    float hv = og * tanhf(cv);
    c[i] = cv;
    h[i] = hv;
    h16[i] = (_Float16)hv;
    hs16[i] = (_Float16)hv;
}

// ---------------------------------------------------------------------------
extern "C" void kernel_launch(void* const* d_in, const int* in_sizes, int n_in,
                              void* d_out, int out_size, void* d_ws, size_t ws_size,
                              hipStream_t stream) {
    const float* x        = (const float*)d_in[0];
    const float* done     = (const float*)d_in[1];
    const float* h0       = (const float*)d_in[2];
    const float* c0       = (const float*)d_in[3];
    const float* conv1_w  = (const float*)d_in[4];
    const float* conv1_b  = (const float*)d_in[5];
    const float* conv2_w  = (const float*)d_in[6];
    const float* conv2_b  = (const float*)d_in[7];
    const float* conv3_w  = (const float*)d_in[8];
    const float* conv3_b  = (const float*)d_in[9];
    const float* fc1_w    = (const float*)d_in[10];
    const float* fc1_b    = (const float*)d_in[11];
    const float* fc2_w    = (const float*)d_in[12];
    const float* fc2_b    = (const float*)d_in[13];
    const float* fc3_w    = (const float*)d_in[14];
    const float* fc3_b    = (const float*)d_in[15];
    const float* w_ih     = (const float*)d_in[16];
    const float* w_hh     = (const float*)d_in[17];
    const float* b_ih     = (const float*)d_in[18];
    const float* b_hh     = (const float*)d_in[19];
    const float* actor_w  = (const float*)d_in[20];
    const float* actor_b  = (const float*)d_in[21];
    const float* critic_w = (const float*)d_in[22];
    const float* critic_b = (const float*)d_in[23];
    float* out = (float*)d_out;

    char* ws = (char*)d_ws;
    size_t off = 0;
    auto alloc = [&](size_t bytes) -> char* {
        char* p = ws + off;
        off += (bytes + 255) & ~(size_t)255;
        return p;
    };

    _Float16* ping  = (_Float16*)alloc((size_t)TB * FEAT * 2);
    _Float16* pong  = (_Float16*)alloc((size_t)TB * FEAT * 2);
    _Float16* wp_fc1  = (_Float16*)alloc((size_t)208 * 32 * 512 * 2);
    _Float16* wp_fc2  = (_Float16*)alloc((size_t)16 * 32 * 512 * 2);
    _Float16* wp_fc3  = (_Float16*)alloc((size_t)16 * 32 * 512 * 2);
    _Float16* wp_ih   = (_Float16*)alloc((size_t)16 * 128 * 512 * 2);
    _Float16* wp_hh   = (_Float16*)alloc((size_t)16 * 128 * 512 * 2);
    _Float16* wp_act  = (_Float16*)alloc((size_t)16 * 2 * 512 * 2);
    _Float16* wp_crit = (_Float16*)alloc((size_t)16 * 2 * 512 * 2);
    _Float16* wp_c1   = (_Float16*)alloc((size_t)21 * 2 * 512 * 2);
    _Float16* wp_c2   = (_Float16*)alloc((size_t)8 * 2 * 512 * 2);
    _Float16* wp_c3   = (_Float16*)alloc((size_t)8 * 2 * 512 * 2);
    float*    bias_l  = (float*)alloc((size_t)G4 * 4);
    float*    G0      = (float*)alloc((size_t)TB * G4 * 4);
    float*    Gt      = (float*)alloc((size_t)BSZ * G4 * 4);
    float*    h32     = (float*)alloc((size_t)BSZ * HID * 4);
    float*    c32     = (float*)alloc((size_t)BSZ * HID * 4);
    _Float16* h16     = (_Float16*)alloc((size_t)BSZ * HID * 2);
    _Float16* hs16    = (_Float16*)alloc((size_t)TB * HID * 2);

    // ---- pack weights to B-fragment layout ----
    pack_b<<<208 * 32, 32, 0, stream>>>(fc1_w, wp_fc1, HID, FEAT, 32);
    pack_b<<<16 * 32, 32, 0, stream>>>(fc2_w, wp_fc2, HID, HID, 32);
    pack_b<<<16 * 32, 32, 0, stream>>>(fc3_w, wp_fc3, HID, HID, 32);
    pack_b<<<16 * 128, 32, 0, stream>>>(w_ih, wp_ih, G4, HID, 128);
    pack_b<<<16 * 128, 32, 0, stream>>>(w_hh, wp_hh, G4, HID, 128);
    pack_b<<<16 * 2, 32, 0, stream>>>(actor_w, wp_act, NACT, HID, 2);
    pack_b<<<16 * 2, 32, 0, stream>>>(critic_w, wp_crit, 1, HID, 2);
    pack_b<<<21 * 2, 32, 0, stream>>>(conv1_w, wp_c1, NCH, NCH * 25, 2);
    pack_b<<<8 * 2, 32, 0, stream>>>(conv2_w, wp_c2, NCH, NCH * 9, 2);
    pack_b<<<8 * 2, 32, 0, stream>>>(conv3_w, wp_c3, NCH, NCH * 9, 2);
    add_vec<<<(G4 + 255) / 256, 256, 0, stream>>>(bias_l, b_ih, b_hh, G4);

    // ---- x -> f16, conv stack ----
    {
        int n = TB * FEAT;
        cvt_f16<<<(n + 255) / 256, 256, 0, stream>>>(ping, x, n);
    }
    conv_wmma<5><<<TB, 512, 0, stream>>>(ping, wp_c1, conv1_b, pong);
    conv_wmma<3><<<TB, 512, 0, stream>>>(pong, wp_c2, conv2_b, ping);
    conv_wmma<3><<<TB, 512, 0, stream>>>(ping, wp_c3, conv3_b, pong);

    // ---- MLP: fc1/fc2/fc3 (leaky) ----
    gemm_wmma<<<dim3(TB / 128, HID / 32), 256, 0, stream>>>(
        pong, wp_fc1, fc1_b, nullptr, 0, nullptr, 0, HID, ping, HID,
        TB, HID, FEAT, 1);
    gemm_wmma<<<dim3(TB / 128, HID / 32), 256, 0, stream>>>(
        ping, wp_fc2, fc2_b, nullptr, 0, nullptr, 0, HID, pong, HID,
        TB, HID, HID, 1);
    gemm_wmma<<<dim3(TB / 128, HID / 32), 256, 0, stream>>>(
        pong, wp_fc3, fc3_b, nullptr, 0, nullptr, 0, HID, ping, HID,
        TB, HID, HID, 1);
    // ping now holds feat f16 [8192][512]

    // ---- hoisted input-side LSTM GEMM for all timesteps ----
    gemm_wmma<<<dim3(TB / 128, G4 / 32), 256, 0, stream>>>(
        ping, wp_ih, bias_l, nullptr, 0, G0, G4, G4, nullptr, 0,
        TB, G4, HID, 0);

    // ---- init recurrent state ----
    copy_f32<<<(BSZ * HID + 255) / 256, 256, 0, stream>>>(h32, h0, BSZ * HID);
    copy_f32<<<(BSZ * HID + 255) / 256, 256, 0, stream>>>(c32, c0, BSZ * HID);

    // ---- sequential scan ----
    for (int t = 0; t < TSTEP; ++t) {
        lstm_mask<<<(BSZ * HID + 255) / 256, 256, 0, stream>>>(
            h32, c32, h16, done + (size_t)t * BSZ);
        gemm_wmma<<<dim3(BSZ / 128, G4 / 32), 256, 0, stream>>>(
            h16, wp_hh, nullptr, G0 + (size_t)t * BSZ * G4, G4,
            Gt, G4, G4, nullptr, 0, BSZ, G4, HID, 0);
        lstm_gates<<<(BSZ * HID + 255) / 256, 256, 0, stream>>>(
            Gt, c32, h32, h16, hs16 + (size_t)t * BSZ * HID);
    }

    // ---- heads -> d_out ----
    float* logits = out;                       // [8192][6]
    float* value  = out + (size_t)TB * NACT;   // [8192][1]
    float* hT     = value + TB;                // [1][128][512]
    float* cT     = hT + BSZ * HID;
    gemm_wmma<<<dim3(TB / 128, 1), 256, 0, stream>>>(
        hs16, wp_act, actor_b, nullptr, 0, logits, NACT, NACT, nullptr, 0,
        TB, 32, HID, 0);
    gemm_wmma<<<dim3(TB / 128, 1), 256, 0, stream>>>(
        hs16, wp_crit, critic_b, nullptr, 0, value, 1, 1, nullptr, 0,
        TB, 32, HID, 0);
    copy_f32<<<(BSZ * HID + 255) / 256, 256, 0, stream>>>(hT, h32, BSZ * HID);
    copy_f32<<<(BSZ * HID + 255) / 256, 256, 0, stream>>>(cT, c32, BSZ * HID);
}
